// UnicornConv1d_54906861912624
// MI455X (gfx1250) — matrix operand order repaired and checked by reading, not compiled, not run on hardware
//
#include <hip/hip_runtime.h>

// Problem constants (match reference)
#define RES      2048
#define SUPPORT  17
#define SEQ      4096
#define MTOT     16384      // 4 * 4096
#define MTILE    128
#define NTILE    128
#define HALO     16
#define MROWS    (MTILE + HALO)   // 144
#define KSTEP    32
#define A_STRIDE 40                    // bf16 elems per LDS row (32 + 8 pad), row = 80B (16B aligned)
#define SA_ELE   (MROWS * A_STRIDE)    // 5760 halves
#define SB_ELE   (NTILE * A_STRIDE)    // 5120 halves
#define STAGE_ELE (SA_ELE + SB_ELE)    // 10880 halves per buffer (21760 B)
#define U_STRIDE 132                   // f32 elems per LDS row of the u-tile
#define SMEM_BYTES 76032               // max(2*21760 = 43520 stage, 144*132*4 = 76032 u-tile)

typedef __attribute__((ext_vector_type(16))) __bf16 v16bf;
typedef __attribute__((ext_vector_type(8)))  __bf16 v8bf;
typedef __attribute__((ext_vector_type(2)))  __bf16 v2bf;
typedef __attribute__((ext_vector_type(8)))  float  v8f;

union V16 { v16bf v; v8bf h[2]; };

__device__ __forceinline__ unsigned int f2bf_bits(float f) {
  union { float f; unsigned int u; } x; x.f = f;
  unsigned int r = x.u + 0x7FFFu + ((x.u >> 16) & 1u);   // round-to-nearest-even
  return r >> 16;
}
__device__ __forceinline__ unsigned int pack2(float a, float b) {
#if __has_builtin(__builtin_amdgcn_cvt_pk_bf16_f32)
  union { v2bf v; unsigned int u; } c;
  c.v = __builtin_amdgcn_cvt_pk_bf16_f32(a, b);          // v_cvt_pk_bf16_f32 (RNE)
  return c.u;
#else
  return f2bf_bits(a) | (f2bf_bits(b) << 16);
#endif
}

__global__ __launch_bounds__(256)
void unicorn_conv1d_kernel(const float* __restrict__ x,
                           const float* __restrict__ Wm,
                           const float* __restrict__ pad,
                           float* __restrict__ out) {
  extern __shared__ unsigned char smem[];
  unsigned short* sStage = (unsigned short*)smem;       // 2 x (A tile + B tile), bf16
  float*          sU     = (float*)smem;                // 144 x U_STRIDE f32 (unions stage)

  const int tid    = threadIdx.x;
  const int lane   = tid & 31;
  const int wv     = tid >> 5;          // wave id 0..7 -> 16-wide N strip
  const int hi     = (lane >> 4) & 1;   // lane half
  const int lrow   = lane & 15;
  const int bx     = blockIdx.x;        // N tile (16 of them)
  const int by     = blockIdx.y;        // M tile (128 of them)
  const int m0     = by * MTILE;
  const int n0base = bx * NTILE;

  // Stage the K-slice [k0, k0+32) of x (with 16-row halo) and W into stage buffer p,
  // converting f32 -> bf16 (RNE) on the fly.
  auto stageTile = [&](int k0, int p) {
    unsigned short* sA = sStage + p * STAGE_ELE;
    unsigned short* sB = sA + SA_ELE;
    for (int idx = tid; idx < MROWS * 2; idx += 256) {
      const int r = idx >> 1, h = idx & 1;
      int grow = m0 - HALO + r; if (grow < 0) grow = 0;  // clamped halo rows are never used
      const float4* src = (const float4*)(x + (size_t)grow * RES + k0 + h * 16);
      const float4 f0 = src[0], f1 = src[1], f2 = src[2], f3 = src[3];
      uint4* dst = (uint4*)(sA + r * A_STRIDE + h * 16);
      dst[0] = make_uint4(pack2(f0.x, f0.y), pack2(f0.z, f0.w),
                          pack2(f1.x, f1.y), pack2(f1.z, f1.w));
      dst[1] = make_uint4(pack2(f2.x, f2.y), pack2(f2.z, f2.w),
                          pack2(f3.x, f3.y), pack2(f3.z, f3.w));
    }
    {   // exactly 256 B-tile tasks
      const int r = tid >> 1, h = tid & 1;
      const float4* src = (const float4*)(Wm + (size_t)(n0base + r) * RES + k0 + h * 16);
      const float4 f0 = src[0], f1 = src[1], f2 = src[2], f3 = src[3];
      uint4* dst = (uint4*)(sB + r * A_STRIDE + h * 16);
      dst[0] = make_uint4(pack2(f0.x, f0.y), pack2(f0.z, f0.w),
                          pack2(f1.x, f1.y), pack2(f1.z, f1.w));
      dst[1] = make_uint4(pack2(f2.x, f2.y), pack2(f2.z, f2.w),
                          pack2(f3.x, f3.y), pack2(f3.z, f3.w));
    }
  };

  v8f acc[9] = {};                      // 9 M-subtiles x (16x16 f32) accumulators

  stageTile(0, 0);                      // prologue: stage first K slice
  int buf = 0;

  for (int k0 = 0; k0 < RES; k0 += KSTEP) {
    __syncthreads();   // stage buffer `buf` ready (compiler emits s_wait_dscnt 0 + barrier)

    const unsigned short* sA = sStage + buf * STAGE_ELE;
    const unsigned short* sB = sA + SA_ELE;

    // ---- issue ALL fragment loads for the current tile first ----
    // B fragment (32x16): lanes 0-15 hold K 0-15, lanes 16-31 hold K 16-31, lane = N
    V16 bf;
    {
      const unsigned short* bbase = sB + (wv * 16 + lrow) * A_STRIDE + hi * 16;
      bf.h[0] = *(const v8bf*)(const void*)(bbase);
      bf.h[1] = *(const v8bf*)(const void*)(bbase + 8);
    }
    // A fragments: lanes 0-15 hold K 0-15, lanes 16-31 hold K 16-31 (matches B), lane = M
    V16 af[9];
    #pragma unroll
    for (int ms = 0; ms < 9; ++ms) {
      const unsigned short* abase = sA + (ms * 16 + lrow) * A_STRIDE + hi * 16;
      af[ms].h[0] = *(const v8bf*)(const void*)(abase);
      af[ms].h[1] = *(const v8bf*)(const void*)(abase + 8);
    }

    // ---- overlap: stage the NEXT K slice into the other buffer while WMMAs run ----
    if (k0 + KSTEP < RES) stageTile(k0 + KSTEP, buf ^ 1);

    // ---- 9 back-to-back WMMAs (independent accumulators, graduated dscnt waits) ----
    #pragma unroll
    for (int ms = 0; ms < 9; ++ms) {
      acc[ms] = __builtin_amdgcn_wmma_f32_16x16x32_bf16(
          false, af[ms].v, false, bf.v, (short)0, acc[ms], false, false);
    }
    buf ^= 1;
  }

  // ---- epilogue: spill u tile (144 x 128 f32) to LDS ----
  __syncthreads();   // all waves done with stage buffers
  {
    const int colLocal = (wv << 4) + lrow;
    #pragma unroll
    for (int ms = 0; ms < 9; ++ms) {
      #pragma unroll
      for (int i = 0; i < 8; ++i) {
        const int r = ms * 16 + hi * 8 + i;    // C/D layout: VGPR i -> M = i + 8*lanehi
        sU[r * U_STRIDE + colLocal] = acc[ms][i];
      }
    }
  }
  __syncthreads();

  // ---- causal tap filter; tap index is uniform per block (128 | 512) ----
  static const int c_n0[4] = {1, 2, 1, 1};
  static const int c_n1[4] = {2, 3, 5, 17};
  const int tap = bx >> 2;
  const int tn0 = c_n0[tap], tn1 = c_n1[tap];
  const float inv = 1.0f / (float)(tn1 - tn0);

  for (int it = 0; it < 64; ++it) {
    const int e   = tid + (it << 8);
    const int row = e >> 7;          // 0..127 within tile
    const int col = e & 127;
    const int m   = m0 + row;
    const int s   = m & (SEQ - 1);   // position within batch
    const int c   = n0base + col;
    float sum = 0.0f;
    for (int n = tn0; n < tn1; ++n) {
      const int sn = s - n;
      sum += (sn >= 0) ? sU[(row + HALO - n) * U_STRIDE + col]
                       : pad[(size_t)(SUPPORT - n + s) * RES + c];
    }
    out[(size_t)m * RES + c] = sum * inv;
  }
}

extern "C" void kernel_launch(void* const* d_in, const int* in_sizes, int n_in,
                              void* d_out, int out_size, void* d_ws, size_t ws_size,
                              hipStream_t stream) {
  (void)in_sizes; (void)n_in; (void)out_size; (void)d_ws; (void)ws_size;
  const float* x   = (const float*)d_in[0];
  const float* Wm  = (const float*)d_in[1];
  const float* pad = (const float*)d_in[2];
  float* out = (float*)d_out;

  dim3 grid(RES / NTILE, MTOT / MTILE);   // (16, 128) = 2048 blocks
  unicorn_conv1d_kernel<<<grid, dim3(256, 1, 1), SMEM_BYTES, stream>>>(x, Wm, pad, out);
}
